// CausalSelfAttention_23957327577286
// MI455X (gfx1250) — matrix operand order repaired and checked by reading, not compiled, not run on hardware
//
#include <hip/hip_runtime.h>

#define Bz 8
#define Tz 1024
#define Cz 768
#define Hz 12
#define Dz 64
#define NEG_INF (-3.0e38f)

typedef __attribute__((ext_vector_type(16))) _Float16 v16h;
typedef __attribute__((ext_vector_type(8)))  float    v8f;
typedef __attribute__((ext_vector_type(4)))  int      v4i;

typedef __attribute__((address_space(1))) v4i g_v4i;  // global
typedef __attribute__((address_space(3))) v4i l_v4i;  // LDS

#define WMMA_F16(a, b, c) \
  __builtin_amdgcn_wmma_f32_16x16x32_f16(false, (a), false, (b), (short)0, (c), false, false)

#if __has_builtin(__builtin_amdgcn_global_load_async_to_lds_b128)
#define HAVE_ASYNC_LDS 1
#else
#define HAVE_ASYNC_LDS 0
#endif

// copy 16B global -> LDS (async on CDNA5: tracked by ASYNCcnt, bypasses VGPRs)
__device__ __forceinline__ void copy16_g2l(const _Float16* g, _Float16* l) {
#if HAVE_ASYNC_LDS
  v4i* gg = (v4i*)const_cast<_Float16*>(g);
  v4i* ll = (v4i*)l;
  __builtin_amdgcn_global_load_async_to_lds_b128((g_v4i*)gg, (l_v4i*)ll, 0, 0);
#else
  *(uint4*)l = *(const uint4*)g;
#endif
}

template <int N>
__device__ __forceinline__ void wait_async() {
#if HAVE_ASYNC_LDS
#if __has_builtin(__builtin_amdgcn_s_wait_asynccnt)
  __builtin_amdgcn_s_wait_asynccnt(N);
#else
  asm volatile("s_wait_asynccnt %0" ::"i"(N) : "memory");
#endif
#endif
}

// ---------------------------------------------------------------------------
// Fragment loaders (wave32; contiguous k = 16*hi + s, same order for A and B)
// A 16x32 f16: lane L -> row = L&15; hi = L>>4.
//   u32[j]   (j=0..7) holds K = hi*16 + 2j, 2j+1
__device__ __forceinline__ v16h load_afrag(const _Float16* rowp, int hi) {
  union { v16h v; unsigned int u[8]; } r;
  const unsigned int* p = (const unsigned int*)rowp;
#pragma unroll
  for (int j = 0; j < 8; ++j) {
    r.u[j]     = p[hi * 8 + j];
  }
  return r.v;
}
// B 32x16 f16: lane L -> col N = L&15; halves h=0..15 hold K = hi*16 + h
// (K contiguous per lane -> read 8 u32 from a K-major row)
__device__ __forceinline__ v16h load_bfrag(const _Float16* colp, int hi) {
  union { v16h v; unsigned int u[8]; } r;
  const unsigned int* p = (const unsigned int*)colp;
#pragma unroll
  for (int j = 0; j < 8; ++j) r.u[j] = p[hi * 8 + j];
  return r.v;
}

// ---------------------------------------------------------------------------
__global__ __launch_bounds__(256) void f32_to_f16_kernel(const float* __restrict__ in,
                                                         _Float16* __restrict__ out, int n) {
  int i = (blockIdx.x * 256 + threadIdx.x) * 4;
  if (i < n) {
    float4 v = *(const float4*)(in + i);
    out[i + 0] = (_Float16)v.x;
    out[i + 1] = (_Float16)v.y;
    out[i + 2] = (_Float16)v.z;
    out[i + 3] = (_Float16)v.w;
  }
}

// Transpose-convert: in [rows][cols] f32 -> out [cols][rows] f16 (LDS tiled)
__global__ __launch_bounds__(256) void transpose_f32_to_f16(const float* __restrict__ in,
                                                            _Float16* __restrict__ out,
                                                            int rows, int cols) {
  __shared__ float tile[32][33];
  int c0 = blockIdx.x * 32, r0 = blockIdx.y * 32;
  int tx = threadIdx.x & 31, ty = threadIdx.x >> 5;
#pragma unroll
  for (int i = 0; i < 32; i += 8)
    tile[ty + i][tx] = in[(size_t)(r0 + ty + i) * cols + c0 + tx];
  __syncthreads();
#pragma unroll
  for (int i = 0; i < 32; i += 8)
    out[(size_t)(c0 + ty + i) * rows + r0 + tx] = (_Float16)tile[tx][ty + i];
}

// ---------------------------------------------------------------------------
// GEMM: out[M][N] = A[M][K] * BT[N][K]^T + bias[N]
// MODE 0: scatter to q/k f16 [B,H,T,D] and v f16 [B,H,D,T].  MODE 1: f32 out.
// Block tile 128x256, BK=32, 256 threads = 8 waves; wave tile 64(M) x 64(N).
// Tiles staged to double-buffered LDS via async global->LDS (ASYNCcnt).
// LDS row stride 40 halves = 80B: 16B aligned for B128 writes, and
// conflict-free for 16 consecutive rows (gcd(20 banks,64)=4 -> distinct).
template <int MODE>
__global__ __launch_bounds__(256) void gemm_kernel(
    const _Float16* __restrict__ Ag, const _Float16* __restrict__ BTg,
    const float* __restrict__ bias, float* __restrict__ outf,
    _Float16* __restrict__ qh, _Float16* __restrict__ kh, _Float16* __restrict__ vt,
    int M, int N, int K) {
  __shared__ _Float16 Asm[2][128 * 40];
  __shared__ _Float16 Bsm[2][256 * 40];

  const int tid = threadIdx.x;
  const int wid = tid >> 5, lane = tid & 31, lq = lane & 15, hi = lane >> 4;
  const int wm = (wid & 1) * 64, wn = (wid >> 1) * 64;
  const int m0 = blockIdx.y * 128, n0 = blockIdx.x * 256;

  const v8f vzero = {0.f, 0.f, 0.f, 0.f, 0.f, 0.f, 0.f, 0.f};
  v8f acc[4][4];
#pragma unroll
  for (int i = 0; i < 4; ++i)
#pragma unroll
    for (int f = 0; f < 4; ++f) acc[i][f] = vzero;

  const int arow = tid >> 1, aks = (tid & 1) * 16;

  auto stage = [&](int pb, int kt) {
    const _Float16* ga = Ag + (size_t)(m0 + arow) * K + kt + aks;
    _Float16* la = &Asm[pb][arow * 40 + aks];
    copy16_g2l(ga, la);
    copy16_g2l(ga + 8, la + 8);
    const _Float16* gb = BTg + (size_t)(n0 + tid) * K + kt;
    _Float16* lb = &Bsm[pb][tid * 40];
#pragma unroll
    for (int c = 0; c < 4; ++c) copy16_g2l(gb + c * 8, lb + c * 8);
  };

  stage(0, 0);
  int pb = 0;
  for (int kt = 0; kt < K; kt += 32, pb ^= 1) {
    const bool more = (kt + 32) < K;
    if (more) {
      stage(pb ^ 1, kt + 32);   // prefetch next tile into other buffer
      wait_async<6>();          // this wave's 6 current-tile asyncs retired
    } else {
      wait_async<0>();
    }
    __syncthreads();

    v16h af[4], bf[4];
#pragma unroll
    for (int i = 0; i < 4; ++i) af[i] = load_afrag(&Asm[pb][(wm + 16 * i + lq) * 40], hi);
#pragma unroll
    for (int f = 0; f < 4; ++f) bf[f] = load_bfrag(&Bsm[pb][(wn + 16 * f + lq) * 40], hi);
#pragma unroll
    for (int i = 0; i < 4; ++i)
#pragma unroll
      for (int f = 0; f < 4; ++f) acc[i][f] = WMMA_F16(af[i], bf[f], acc[i][f]);
    __syncthreads();
  }

  // epilogue
#pragma unroll
  for (int i = 0; i < 4; ++i) {
#pragma unroll
    for (int f = 0; f < 4; ++f) {
      const int n = n0 + wn + f * 16 + lq;
      const float bv = bias[n];
#pragma unroll
      for (int r = 0; r < 8; ++r) {
        const int m = m0 + wm + i * 16 + r + 8 * hi;
        const float val = acc[i][f][r] + bv;
        if (MODE == 1) {
          outf[(size_t)m * N + n] = val;
        } else {
          const int g = n / Cz;  // 0=q 1=k 2=v
          const int cc = n - g * Cz;
          const int h = cc >> 6, d = cc & 63;
          const int b = m >> 10, t = m & 1023;
          if (g == 2) {  // V transposed: [B,H,D,T] for contiguous-K PV fragments
            vt[(size_t)((b * Hz + h) * Dz + d) * Tz + t] = (_Float16)val;
          } else {
            _Float16* dst = (g == 0) ? qh : kh;
            dst[(size_t)((b * Hz + h) * Tz + t) * Dz + d] = (_Float16)val;
          }
        }
      }
    }
  }
}

// ---------------------------------------------------------------------------
// Flash attention: 1 wave = 16 query rows; 8 waves/block -> 128 q rows.
// grid.x = B*H*(T/128); per-wave online softmax, kv step = 32.
// 1/sqrt(D)=0.125=2^-3 is exact in f16 -> folded into Q at load (no muls).
__global__ __launch_bounds__(256) void attn_kernel(const _Float16* __restrict__ qh,
                                                   const _Float16* __restrict__ kh,
                                                   const _Float16* __restrict__ vt,
                                                   _Float16* __restrict__ yh) {
  __shared__ _Float16 psm[8][16][34];  // per-wave P bounce (D-layout -> A-layout)

  const int tid = threadIdx.x;
  const int wid = tid >> 5, lane = tid & 31, lq = lane & 15, hi = lane >> 4;
  const int bh = blockIdx.x >> 3;
  const int qbase = (blockIdx.x & 7) * 128 + wid * 16;

  const _Float16* qp = qh + (size_t)bh * Tz * Dz;
  const _Float16* kp = kh + (size_t)bh * Tz * Dz;
  const _Float16* vp = vt + (size_t)bh * Dz * Tz;  // [D][T]

  v16h qf0 = load_afrag(qp + (qbase + lq) * Dz + 0, hi);   // d 0..31
  v16h qf1 = load_afrag(qp + (qbase + lq) * Dz + 32, hi);  // d 32..63
  {
    const _Float16 sc = (_Float16)0.125f;  // exact power of two
#pragma unroll
    for (int j = 0; j < 16; ++j) { qf0[j] *= sc; qf1[j] *= sc; }
  }

  const v8f vzero = {0.f, 0.f, 0.f, 0.f, 0.f, 0.f, 0.f, 0.f};
  v8f oacc[4];
#pragma unroll
  for (int db = 0; db < 4; ++db) oacc[db] = vzero;
  float ms[8], ls[8];
#pragma unroll
  for (int r = 0; r < 8; ++r) { ms[r] = NEG_INF; ls[r] = 0.f; }

  auto kv_step = [&](int kv0, bool domask) {
    // S = Q @ K^T for 32 keys, as two 16x16 f32 tiles
    v8f s0 = vzero, s1 = vzero;
    {
      v16h k00 = load_bfrag(kp + (kv0 + lq) * Dz + 0, hi);
      v16h k01 = load_bfrag(kp + (kv0 + lq) * Dz + 32, hi);
      v16h k10 = load_bfrag(kp + (kv0 + 16 + lq) * Dz + 0, hi);
      v16h k11 = load_bfrag(kp + (kv0 + 16 + lq) * Dz + 32, hi);
      s0 = WMMA_F16(qf0, k00, s0);
      s0 = WMMA_F16(qf1, k01, s0);
      s1 = WMMA_F16(qf0, k10, s1);
      s1 = WMMA_F16(qf1, k11, s1);
    }
    // online softmax (row = qbase + r + 8*hi, col = kv0 + lq (+16))
    const int c0 = kv0 + lq, c1 = c0 + 16;
#pragma unroll
    for (int r = 0; r < 8; ++r) {
      float x0 = s0[r];
      float x1 = s1[r];
      if (domask) {
        const int row = qbase + r + 8 * hi;
        x0 = (c0 <= row) ? x0 : NEG_INF;
        x1 = (c1 <= row) ? x1 : NEG_INF;
      }
      float mx = fmaxf(x0, x1);
#pragma unroll
      for (int sh = 1; sh < 16; sh <<= 1) mx = fmaxf(mx, __shfl_xor(mx, sh, 32));
      const float mnew = fmaxf(ms[r], mx);
      const float alpha = __expf(ms[r] - mnew);
      const float p0 = __expf(x0 - mnew);
      const float p1 = __expf(x1 - mnew);
      float rs = p0 + p1;
#pragma unroll
      for (int sh = 1; sh < 16; sh <<= 1) rs += __shfl_xor(rs, sh, 32);
      ls[r] = ls[r] * alpha + rs;
      ms[r] = mnew;
#pragma unroll
      for (int db = 0; db < 4; ++db) oacc[db][r] *= alpha;
      psm[wid][hi * 8 + r][lq]      = (_Float16)p0;  // D-layout -> plain [row][col]
      psm[wid][hi * 8 + r][16 + lq] = (_Float16)p1;
    }
    // reload P as A fragment (wave-private LDS; DS ops are in-order per wave)
    const v16h pf = load_afrag(&psm[wid][lq][0], hi);
    // O += P @ V   (V^T rows are d-columns: contiguous-K B fragments)
#pragma unroll
    for (int db = 0; db < 4; ++db) {
      v16h vf = load_bfrag(vp + (size_t)(db * 16 + lq) * Tz + kv0, hi);
      oacc[db] = WMMA_F16(pf, vf, oacc[db]);
    }
  };

  const int kvend = qbase + 16;          // causal: max col = qbase+15
  const int nfull = ((qbase + 1) >> 5);  // tiles with kv0+31 <= qbase: mask-free
  int kv0 = 0;
  for (; kv0 < nfull * 32; kv0 += 32) {
    // prefetch next K/V tiles while this one computes (global_prefetch_b8)
    __builtin_prefetch(kp + (size_t)(kv0 + 32 + lq) * Dz, 0, 1);
    __builtin_prefetch(kp + (size_t)(kv0 + 48 + lq) * Dz, 0, 1);
    __builtin_prefetch(vp + (size_t)lane * Tz + kv0 + 32, 0, 1);
    __builtin_prefetch(vp + (size_t)(lane + 32) * Tz + kv0 + 32, 0, 1);
    kv_step(kv0, false);
  }
  for (; kv0 < kvend; kv0 += 32) kv_step(kv0, true);

  // write y (f16, [B,T,C] layout) with 1/l normalization
  const int b = bh / Hz, h = bh % Hz;
#pragma unroll
  for (int db = 0; db < 4; ++db) {
#pragma unroll
    for (int r = 0; r < 8; ++r) {
      const int t = qbase + r + 8 * hi;
      const int col = h * Dz + db * 16 + lq;
      yh[(size_t)(b * Tz + t) * Cz + col] = (_Float16)(oacc[db][r] / ls[r]);
    }
  }
}

// ---------------------------------------------------------------------------
extern "C" void kernel_launch(void* const* d_in, const int* in_sizes, int n_in,
                              void* d_out, int out_size, void* d_ws, size_t ws_size,
                              hipStream_t stream) {
  const float* x      = (const float*)d_in[0];
  const float* w_attn = (const float*)d_in[1];
  const float* b_attn = (const float*)d_in[2];
  const float* w_proj = (const float*)d_in[3];
  const float* b_proj = (const float*)d_in[4];

  char* ws = (char*)d_ws;
  size_t off = 0;
  auto alloc = [&](size_t bytes) {
    char* p = ws + off;
    off += (bytes + 255) & ~(size_t)255;
    return p;
  };
  const size_t nBTC = (size_t)Bz * Tz * Cz;  // 6,291,456
  _Float16* xh     = (_Float16*)alloc(nBTC * 2);
  _Float16* wqkvT  = (_Float16*)alloc((size_t)Cz * 3 * Cz * 2);
  _Float16* wprojT = (_Float16*)alloc((size_t)Cz * Cz * 2);
  _Float16* qhd    = (_Float16*)alloc(nBTC * 2);
  _Float16* khd    = (_Float16*)alloc(nBTC * 2);
  _Float16* vtd    = (_Float16*)alloc(nBTC * 2);  // transposed [B,H,D,T]
  _Float16* yh     = xh;  // x no longer needed after QKV GEMM

  // conversions
  f32_to_f16_kernel<<<(int)(nBTC / 1024), 256, 0, stream>>>(x, xh, (int)nBTC);
  transpose_f32_to_f16<<<dim3(3 * Cz / 32, Cz / 32), 256, 0, stream>>>(w_attn, wqkvT, Cz, 3 * Cz);
  transpose_f32_to_f16<<<dim3(Cz / 32, Cz / 32), 256, 0, stream>>>(w_proj, wprojT, Cz, Cz);

  // qkv = x @ w_attn + b_attn -> q/k [B,H,T,D], v [B,H,D,T] (f16)
  gemm_kernel<0><<<dim3(3 * Cz / 256, Bz * Tz / 128), 256, 0, stream>>>(
      xh, wqkvT, b_attn, nullptr, qhd, khd, vtd, Bz * Tz, 3 * Cz, Cz);

  // flash attention -> yh f16 [B,T,C]
  attn_kernel<<<Bz * Hz * (Tz / 128), 256, 0, stream>>>(qhd, khd, vtd, yh);

  // out = y @ w_proj + b_proj (f32)
  gemm_kernel<1><<<dim3(Cz / 256, Bz * Tz / 128), 256, 0, stream>>>(
      yh, wprojT, b_proj, (float*)d_out, nullptr, nullptr, nullptr, Bz * Tz, Cz, Cz);
}
